// SE3DivergenceFreeVectorField_75892072120802
// MI455X (gfx1250) — matrix operand order repaired and checked by reading, not compiled, not run on hardware
//
#include <hip/hip_runtime.h>
#include <hip/hip_bf16.h>
#include <math.h>

// ---------------------------------------------------------------------------
// SE3 divergence-free vector field — live-path only.
// Only potential[:, :3] reaches the output, so per edge:
//   emb(10) -> h = silu(emb @ W1s) (WMMA f16 16x16x32, K padded to 32)
//   y = h @ W2eff(64x24)          (WMMA f16 16x16x32, 2 K-steps, 2 N-tiles)
//   out_w = sum_u x_src[u] * y[u*3+w], p = w_cut*out, curl-scatter atomics.
// 16 edges per wave tile; wave32; A/B fragments use contiguous K order (k = 16*half + s).
// Transcendentals use the HW v_exp_f32 / v_rcp_f32 / v_sqrt_f32 units (TRANS
// ops co-execute with XDL WMMA, and we truncate to f16 anyway).
// ---------------------------------------------------------------------------

typedef __attribute__((ext_vector_type(16))) _Float16 v16h;
typedef __attribute__((ext_vector_type(8)))  float    v8f;

union Frag16 {
    v16h      v;
    _Float16  e[16];
    unsigned  u[8];
};

#define WAVES_PER_BLOCK 8
#define TILES_PER_WAVE  8
#define BLOCK_THREADS   (WAVES_PER_BLOCK * 32)

__device__ __forceinline__ float fast_exp(float x) {
    // e^x = 2^(x * log2(e)) -> v_mul + v_exp_f32
    return __builtin_amdgcn_exp2f(x * 1.4426950408889634f);
}
__device__ __forceinline__ float fast_rcp(float x) {
    return __builtin_amdgcn_rcpf(x);     // v_rcp_f32
}

__device__ __forceinline__ int a_kbase(int v, int half) {
    // 16-bit A-matrix 16x32, contiguous K order: VGPR v, lane-half -> starting K index 16*half + 2*v
    return 2 * v + 16 * half;
}

__device__ __forceinline__ v8f wmma_f16(const v16h& a, const v16h& b, v8f c) {
    return __builtin_amdgcn_wmma_f32_16x16x32_f16(
        /*neg_a=*/false, a, /*neg_b=*/false, b,
        /*c_mod=*/(short)0, c, /*reuse_a=*/false, /*reuse_b=*/false);
}

__global__ void se3_zero_kernel(float* out, int n) {
    int i = blockIdx.x * blockDim.x + threadIdx.x;
    if (i < n) out[i] = 0.0f;
}

__global__ __launch_bounds__(BLOCK_THREADS)
void se3_edge_kernel(const float* __restrict__ xfeat,   // (B*N, 8)
                     const float* __restrict__ pos,     // (B*N, 3)
                     const int*   __restrict__ esrc,    // (E)
                     const int*   __restrict__ edst,    // (E)
                     const float* __restrict__ W1,      // (10, 64)
                     const float* __restrict__ W2,      // (64, 384)
                     float*       __restrict__ out,     // (B*N, 3) curl accum
                     int numTiles)
{
    __shared__ _Float16 s_emb[WAVES_PER_BLOCK][16 * 32];  // A for matmul1
    __shared__ _Float16 s_h  [WAVES_PER_BLOCK][16 * 64];  // A for matmul2
    __shared__ float    s_y  [WAVES_PER_BLOCK][16 * 24];  // matmul2 result

    const int lane = threadIdx.x & 31;
    const int wid  = threadIdx.x >> 5;
    const int m    = lane & 15;        // row / column within 16x16 tile
    const int half = lane >> 4;        // lane-half selects K / M offsets

    const int waveGlobal = blockIdx.x * WAVES_PER_BLOCK + wid;

    const float INV_SQRT10 = 0.316227766016838f;                  // fold into W1
    const float W2SCALE    = (1.0f / 8.0f) * 0.353553390593274f;  // 1/sqrt(64)*1/sqrt(8)

    // ---- Build B fragments for W1s (10x64, K padded to 32; 4 N-tiles) ----
    Frag16 bW1[4];
#pragma unroll
    for (int n = 0; n < 4; ++n) {
        const int col = 16 * n + m;
#pragma unroll
        for (int v = 0; v < 8; ++v) {
            const int k0 = 2 * v + 16 * half;
            float f0 = (k0     < 10) ? W1[k0 * 64 + col]       * INV_SQRT10 : 0.0f;
            float f1 = (k0 + 1 < 10) ? W1[(k0 + 1) * 64 + col] * INV_SQRT10 : 0.0f;
            bW1[n].e[2 * v]     = (_Float16)f0;
            bW1[n].e[2 * v + 1] = (_Float16)f1;
        }
    }

    // ---- Build B fragments for W2eff (64x24 gathered cols; pad N to 32) ----
    // Live columns of W2: j = u*3 + w  ->  W2 column u*32 + w  (u<8, w<3)
    Frag16 bW2[2][2];  // [N-tile][K-step]
#pragma unroll
    for (int t = 0; t < 2; ++t) {
        const int  j     = 16 * t + m;
        const bool valid = (j < 24);
        const int  col   = valid ? (j / 3) * 32 + (j % 3) : 0;
#pragma unroll
        for (int s = 0; s < 2; ++s) {
#pragma unroll
            for (int v = 0; v < 8; ++v) {
                const int k0 = 32 * s + 2 * v + 16 * half;
                float f0 = valid ? W2[k0 * 384 + col]       * W2SCALE : 0.0f;
                float f1 = valid ? W2[(k0 + 1) * 384 + col] * W2SCALE : 0.0f;
                bW2[t][s].e[2 * v]     = (_Float16)f0;
                bW2[t][s].e[2 * v + 1] = (_Float16)f1;
            }
        }
    }

    const float CEMB     = 1.14136f * 7.38905609893065f * 3.16227766016838f; // 1.14136*e^2*sqrt(10)
    const float INV_STEP = 11.0f / 3.0f;

    for (int it = 0; it < TILES_PER_WAVE; ++it) {
        const int tile = waveGlobal * TILES_PER_WAVE + it;
        if (tile >= numTiles) break;

        // ---- Phase 1: lanes 0..15 each own one edge: emb, w_cut, x_src ----
        float x[8];
        float wcut = 0.0f;
        int   dst  = 0;
        if (half == 0) {
            const long e   = (long)tile * 16 + m;
            const int  src = esrc[e];
            dst            = edst[e];
            if (it + 1 < TILES_PER_WAVE) {           // gfx1250 global_prefetch_b8
                __builtin_prefetch(&esrc[e + 16], 0, 1);
                __builtin_prefetch(&edst[e + 16], 0, 1);
            }
            const float* ps = pos + 3l * src;
            const float* pd = pos + 3l * dst;
            const float ex = ps[0] - pd[0];
            const float ey = ps[1] - pd[1];
            const float ez = ps[2] - pd[2];
            const float r  = __builtin_amdgcn_sqrtf(ex * ex + ey * ey + ez * ez + 1e-18f);

            _Float16* arow = &s_emb[wid][m * 32];
#pragma unroll
            for (int b = 0; b < 10; ++b) {
                const float center = (b + 1) * (3.0f / 11.0f);
                const float d  = (r - center) * INV_STEP;
                const bool inside = fabsf(d) < 1.0f;
                const float y  = inside ? fmaxf(1.0f - d * d, 1e-7f) : 1.0f;
                const float em = inside ? CEMB * fast_exp(-fast_rcp(y)) : 0.0f;
                arow[b] = (_Float16)em;
            }
#pragma unroll
            for (int b = 10; b < 32; ++b) arow[b] = (_Float16)0.0f;

            const float t10 = 10.0f * (1.0f - r * (1.0f / 3.0f));
            wcut = (t10 > 0.0f) ? fast_exp(-fast_rcp(t10)) : 0.0f;

            const float* xr = xfeat + 8l * src;
#pragma unroll
            for (int u = 0; u < 8; ++u) x[u] = xr[u];
        }
        asm volatile("s_wait_dscnt 0x0" ::: "memory");
        __builtin_amdgcn_wave_barrier();

        // ---- Matmul 1: h = silu(emb @ W1s)  (4 WMMAs) ----
        Frag16 A;
#pragma unroll
        for (int v = 0; v < 8; ++v) {
            const int kb = a_kbase(v, half);
            A.u[v] = *(const unsigned*)&s_emb[wid][m * 32 + kb];
        }
#pragma unroll
        for (int n = 0; n < 4; ++n) {
            v8f c = {};
            c = wmma_f16(A.v, bW1[n].v, c);
            const int ncol = 16 * n + m;
#pragma unroll
            for (int v = 0; v < 8; ++v) {
                const float z = c[v];
                const float hs = z * fast_rcp(1.0f + fast_exp(-z));   // silu
                s_h[wid][(v + 8 * half) * 64 + ncol] = (_Float16)hs;
            }
        }
        asm volatile("s_wait_dscnt 0x0" ::: "memory");
        __builtin_amdgcn_wave_barrier();

        // ---- Matmul 2: y = h @ W2eff  (4 WMMAs, K=64 in 2 steps) ----
        Frag16 H0, H1;
#pragma unroll
        for (int v = 0; v < 8; ++v) {
            const int kb = a_kbase(v, half);
            H0.u[v] = *(const unsigned*)&s_h[wid][m * 64 + kb];
            H1.u[v] = *(const unsigned*)&s_h[wid][m * 64 + 32 + kb];
        }
#pragma unroll
        for (int t = 0; t < 2; ++t) {
            v8f y = {};
            y = wmma_f16(H0.v, bW2[t][0].v, y);
            y = wmma_f16(H1.v, bW2[t][1].v, y);
            const int j = 16 * t + m;
            if (j < 24) {
#pragma unroll
                for (int v = 0; v < 8; ++v)
                    s_y[wid][(v + 8 * half) * 24 + j] = y[v];
            }
        }
        asm volatile("s_wait_dscnt 0x0" ::: "memory");
        __builtin_amdgcn_wave_barrier();

        // ---- Phase 3: contract with x_src, apply w_cut + curl, scatter ----
        if (half == 0) {
            const float* yr = &s_y[wid][m * 24];
            float o0 = 0.0f, o1 = 0.0f, o2 = 0.0f;
#pragma unroll
            for (int u = 0; u < 8; ++u) {
                o0 += x[u] * yr[u * 3 + 0];
                o1 += x[u] * yr[u * 3 + 1];
                o2 += x[u] * yr[u * 3 + 2];
            }
            const float p0 = wcut * o0;
            const float p1 = wcut * o1;
            const float p2 = wcut * o2;
            float* o = out + 3l * dst;
            atomicAdd(o + 0, p2 - p1);   // curl_x = psi_z - psi_y
            atomicAdd(o + 1, p0 - p2);   // curl_y = psi_x - psi_z
            atomicAdd(o + 2, p1 - p0);   // curl_z = psi_y - psi_x
        }
    }
}

extern "C" void kernel_launch(void* const* d_in, const int* in_sizes, int n_in,
                              void* d_out, int out_size, void* d_ws, size_t ws_size,
                              hipStream_t stream) {
    const float* xfeat = (const float*)d_in[0];  // node_features (B*N*8)
    const float* pos   = (const float*)d_in[1];  // pos           (B*N*3)
    const int*   esrc  = (const int*)  d_in[2];  // edge_src      (E)
    const int*   edst  = (const int*)  d_in[3];  // edge_dst      (E)
    const float* W1    = (const float*)d_in[4];  // (10, 64)
    const float* W2    = (const float*)d_in[5];  // (64, 384)
    float* out = (float*)d_out;                  // curl (B*N*3) f32

    const int E        = in_sizes[2];
    const int numTiles = E / 16;                 // E is a multiple of 16

    // Zero the curl accumulator (we scatter-add into it).
    se3_zero_kernel<<<(out_size + 255) / 256, 256, 0, stream>>>(out, out_size);

    const int tilesPerBlock = WAVES_PER_BLOCK * TILES_PER_WAVE;
    const int blocks = (numTiles + tilesPerBlock - 1) / tilesPerBlock;
    se3_edge_kernel<<<blocks, BLOCK_THREADS, 0, stream>>>(
        xfeat, pos, esrc, edst, W1, W2, out, numTiles);
}
